// WindowAttention_78520592105956
// MI455X (gfx1250) — matrix operand order repaired and checked by reading, not compiled, not run on hardware
//
#include <hip/hip_runtime.h>
#include <hip/hip_bf16.h>

// ---------------------------------------------------------------------------
// Swin window attention on gfx1250 (MI455X), wave32 + v_wmma_f32_16x16x32_f16
// + Tensor Data Mover (tensor_load_to_lds) for staging the input window.
// One workgroup (256 threads = 8 waves) per window.
// ---------------------------------------------------------------------------

typedef _Float16 v16h __attribute__((ext_vector_type(16)));
typedef _Float16 v8h  __attribute__((ext_vector_type(8)));
typedef _Float16 v4h  __attribute__((ext_vector_type(4)));
typedef float    v8f  __attribute__((ext_vector_type(8)));
typedef float    v4f  __attribute__((ext_vector_type(4)));
typedef unsigned v4u  __attribute__((ext_vector_type(4)));
typedef unsigned v8u  __attribute__((ext_vector_type(8)));

#define DIMC      128
#define HEADS     4
#define HEAD_DIM  32
#define NTOK      49
#define NPAD      64
#define NW        64
#define SCALE_Q   0.17677669529663687f   // 32^-0.5
#define THREE_C   384
#define VSTRIDE   72                     // transposed-V row stride (halves)
#define NEG_BIG   -60000.0f              // finite in f16; exp() underflows to 0

// LDS layout in halves:
#define LDS_XS    0                      // 64x128 f16 (x, later attn output)
#define LDS_QS    8192                   // 64x128 f16
#define LDS_KS    16384                  // 64x128 f16
#define LDS_VST   24576                  // 128x72 f16 (V transposed)
#define LDS_SC    33792                  // [4][64][64] f16 scores/probs
#define LDS_TOTAL 50176                  // 100352 bytes
#define XSTAGE_BYTE_OFF (LDS_SC * 2)     // f32 staging overlays sc region

// ---- WMMA fragment loaders (contiguous k = 16h + s, same order for A and B) ---------------------

// A-matrix 16x32 f16: lane holds row m; K contiguous 16 per lane half.
__device__ __forceinline__ v16h load_a_f16(const _Float16* base, int m,
                                           int stride, int kb, int lane) {
  const int hi = (lane & 16) ? 1 : 0;
  const int k0 = kb + (hi ? 16 : 0);
  const int k1 = k0 + 8;
  v8h lo = *(const v8h*)(base + m * stride + k0);
  v8h hh = *(const v8h*)(base + m * stride + k1);
  return __builtin_shufflevector(lo, hh, 0, 1, 2, 3, 4, 5, 6, 7,
                                 8, 9, 10, 11, 12, 13, 14, 15);
}

// B-matrix 32x16 f16, column n contiguous in memory, 32-byte aligned rows.
__device__ __forceinline__ v16h load_b_rowmajor(const _Float16* base, int n,
                                                int stride, int kb, int lane) {
  const int k0 = kb + ((lane & 16) ? 16 : 0);
  return *(const v16h*)(base + n * stride + k0);
}

// Same, but rows only 16-byte aligned (stride 72 halves): two v8h loads.
__device__ __forceinline__ v16h load_b_rowmajor16(const _Float16* base, int n,
                                                  int stride, int kb, int lane) {
  const int k0 = kb + ((lane & 16) ? 16 : 0);
  v8h lo = *(const v8h*)(base + n * stride + k0);
  v8h hh = *(const v8h*)(base + n * stride + k0 + 8);
  return __builtin_shufflevector(lo, hh, 0, 1, 2, 3, 4, 5, 6, 7,
                                 8, 9, 10, 11, 12, 13, 14, 15);
}

__device__ __forceinline__ v8f wmma_f16(v16h a, v16h b, v8f c) {
  return __builtin_amdgcn_wmma_f32_16x16x32_f16(false, a, false, b,
                                                (short)0, c, false, false);
}

// ---- Prep kernel: weights -> f16, gather relative-position bias ------------

__global__ __launch_bounds__(256) void swin_prep_kernel(
    const float* __restrict__ qkv_w,     // (384,128)
    const float* __restrict__ proj_w,    // (128,128)
    const float* __restrict__ rpb_table, // (169,4)
    const int*   __restrict__ rp_index,  // (49,49)
    _Float16* __restrict__ qkvw_h,
    _Float16* __restrict__ projw_h,
    float*    __restrict__ bias) {       // (4,49,49)
  const int i = blockIdx.x * 256 + threadIdx.x;
  if (i < THREE_C * DIMC) qkvw_h[i] = (_Float16)qkv_w[i];
  if (i < DIMC * DIMC)    projw_h[i] = (_Float16)proj_w[i];
  if (i < HEADS * NTOK * NTOK) {
    const int h = i / (NTOK * NTOK);
    const int p = i % (NTOK * NTOK);
    bias[i] = rpb_table[rp_index[p] * HEADS + h];
  }
}

// ---- Main kernel: one window per workgroup ---------------------------------

__global__ __launch_bounds__(256) void swin_attn_kernel(
    const float* __restrict__ x,       // (4096,49,128)
    const float* __restrict__ mask,    // (64,49,49)
    const float* __restrict__ qkv_b,   // (384)
    const float* __restrict__ proj_b,  // (128)
    const _Float16* __restrict__ qkvw, // (384,128) f16
    const _Float16* __restrict__ projw,// (128,128) f16
    const float* __restrict__ bias,    // (4,49,49) f32
    float* __restrict__ out) {         // (4096,49,128)
  extern __shared__ _Float16 lds[];
  _Float16* xs  = lds + LDS_XS;
  _Float16* qs  = lds + LDS_QS;
  _Float16* ks  = lds + LDS_KS;
  _Float16* vst = lds + LDS_VST;
  _Float16* sc  = lds + LDS_SC;

  const int tid   = threadIdx.x;
  const int lane  = tid & 31;
  const int wv    = tid >> 5;
  const int l15   = lane & 15;
  const int rbase = (lane & 16) ? 8 : 0;   // C/D row offset per lane group
  const int b     = blockIdx.x;

  // Warm L2/L1 with the weight tiles this wave will touch.
  __builtin_prefetch(qkvw + wv * 6144, 0, 0);

  // ---- Phase 1a: TDM DMA of x window (49x128 f32) into LDS staging --------
  if (wv == 0) {
    const unsigned long long ga =
        (unsigned long long)(const void*)(x + (size_t)b * (NTOK * DIMC));
    // D# group 0: count=1 | lds_addr | global_addr[56:0] | type=2
    v4u g0;
    g0[0] = 1u;                                  // count=1, user descriptor
    g0[1] = (unsigned)XSTAGE_BYTE_OFF;           // lds_addr (bytes)
    g0[2] = (unsigned)(ga & 0xffffffffu);        // global_addr[31:0]
    g0[3] = (unsigned)((ga >> 32) & 0x1ffffffu)  // global_addr[56:32]
            | (2u << 30);                        // type = image
    // D# group 1: data_size=4B, tensor 128x49, tile 128x49, stride 128
    v8u g1;
    g1[0] = (2u << 16);                          // data_size = 4 bytes
    g1[1] = (unsigned)(DIMC) << 16;              // tensor_dim0 low (bits 79:48)
    g1[2] = (unsigned)(NTOK) << 16;              // dim0 hi=0 | tensor_dim1 low
    g1[3] = (unsigned)(DIMC) << 16;              // dim1 hi=0 | tile_dim0=128
    g1[4] = (unsigned)(NTOK);                    // tile_dim1=49, tile_dim2=0
    g1[5] = (unsigned)(DIMC);                    // tensor_dim0_stride = 128
    g1[6] = 0u;                                  // stride hi | dim1_stride low
    g1[7] = 0u;
    asm volatile("tensor_load_to_lds %0, %1" :: "s"(g0), "s"(g1) : "memory");
    __builtin_amdgcn_s_wait_tensorcnt(0);
  }
  __syncthreads();

  // ---- Phase 1b: staged f32 -> f16 xs, 8 elements/iter (rows >=49 zeroed) -
  {
    const float* xstage = (const float*)((const char*)lds + XSTAGE_BYTE_OFF);
    for (int i = tid; i < (NPAD * DIMC) / 8; i += 256) {
      const int row = i >> 4;            // 16 chunks of 8 per 128-wide row
      const int col = (i & 15) * 8;
      v4f a0 = {}, a1 = {};
      if (row < NTOK) {
        a0 = *(const v4f*)(xstage + row * DIMC + col);
        a1 = *(const v4f*)(xstage + row * DIMC + col + 4);
      }
      v8h o;
#pragma unroll
      for (int e = 0; e < 4; ++e) { o[e] = (_Float16)a0[e]; o[4 + e] = (_Float16)a1[e]; }
      *(v8h*)(xs + row * DIMC + col) = o;
    }
  }
  __syncthreads();

  // ---- Phase 2: QKV GEMM (64x128 @ 128x384), 96 tiles over 8 waves --------
  for (int i = 0; i < 12; ++i) {
    const int t  = wv * 12 + i;
    const int tm = t & 3;
    const int tn = t >> 2;               // 0..23
    const int m  = tm * 16 + l15;
    const int n  = tn * 16 + l15;
    v8f acc = {};
#pragma unroll
    for (int kk = 0; kk < 4; ++kk) {
      v16h a  = load_a_f16(xs, m, DIMC, kk * 32, lane);
      v16h bf = load_b_rowmajor(qkvw, n, DIMC, kk * 32, lane);
      acc = wmma_f16(a, bf, acc);
    }
    const float bv  = qkv_b[n];
    const float scl = (n < DIMC) ? SCALE_Q : 1.0f;
    if (n < 2 * DIMC) {
      _Float16* dst = (n < DIMC) ? qs : ks;
      const int coln = n & (DIMC - 1);
#pragma unroll
      for (int r = 0; r < 8; ++r) {
        const int mr = tm * 16 + r + rbase;
        dst[mr * DIMC + coln] = (_Float16)((acc[r] + bv) * scl);
      }
    } else {
      // V stored transposed: vst[col][token], stride 72 halves -> b128 store.
      const int coln = n - 2 * DIMC;
#pragma unroll
      for (int r = 0; r < 8; ++r) {
        const int mr = tm * 16 + r + rbase;
        vst[coln * VSTRIDE + mr] = (_Float16)(acc[r] + bv);
      }
    }
  }
  __syncthreads();

  // ---- Phase 3: scores Q*K^T + bias + mask; pads get NEG_BIG --------------
  {
    const float* mrow = mask + (size_t)(b & (NW - 1)) * NTOK * NTOK;
    for (int i = 0; i < 8; ++i) {
      const int t   = wv * 8 + i;
      const int h   = t >> 4;
      const int rem = t & 15;
      const int tm  = rem & 3;
      const int tn  = rem >> 2;
      const int m   = tm * 16 + l15;
      const int n   = tn * 16 + l15;
      v8f acc = {};
      v16h a  = load_a_f16(qs, m, DIMC, h * HEAD_DIM, lane);
      v16h bf = load_b_rowmajor(ks, n, DIMC, h * HEAD_DIM, lane);
      acc = wmma_f16(a, bf, acc);
      const bool nvalid = (n < NTOK);
#pragma unroll
      for (int r = 0; r < 8; ++r) {
        const int mr  = tm * 16 + r + rbase;
        const int m49 = (mr < NTOK) ? mr : NTOK - 1;
        const float s = nvalid
            ? acc[r] + bias[h * NTOK * NTOK + m49 * NTOK + n] + mrow[m49 * NTOK + n]
            : NEG_BIG;
        sc[h * (NPAD * NPAD) + mr * NPAD + n] = (_Float16)s;
      }
    }
  }
  __syncthreads();

  // ---- Phase 4: branch-free vectorized softmax (thread = [head,row]) -------
  // Pad columns hold NEG_BIG -> exp underflows to exactly 0; no masking needed.
  {
    const int h = tid >> 6;
    const int m = tid & 63;
    _Float16* row = sc + h * (NPAD * NPAD) + m * NPAD;
    v8h c[8];
#pragma unroll
    for (int i = 0; i < 8; ++i) c[i] = *(const v8h*)(row + i * 8);
    float mx = NEG_BIG;
#pragma unroll
    for (int i = 0; i < 8; ++i)
#pragma unroll
      for (int e = 0; e < 8; ++e) {
        const float v = (float)c[i][e];
        mx = (v > mx) ? v : mx;
      }
    float s = 0.0f;
#pragma unroll
    for (int i = 0; i < 8; ++i)
#pragma unroll
      for (int e = 0; e < 8; ++e) {
        const float ev = __expf((float)c[i][e] - mx);
        s += ev;
        c[i][e] = (_Float16)ev;
      }
    const float inv = 1.0f / s;
#pragma unroll
    for (int i = 0; i < 8; ++i) {
#pragma unroll
      for (int e = 0; e < 8; ++e)
        c[i][e] = (_Float16)((float)c[i][e] * inv);
      *(v8h*)(row + i * 8) = c[i];
    }
  }
  __syncthreads();

  // ---- Phase 5: P @ V -> xs (reused as attention output), 32 tiles ---------
  for (int i = 0; i < 4; ++i) {
    const int t   = wv * 4 + i;
    const int h   = t >> 3;
    const int rem = t & 7;
    const int tm  = rem & 3;
    const int tn  = rem >> 2;            // 0..1 (hd halves)
    const int m   = tm * 16 + l15;
    const int col = h * HEAD_DIM + tn * 16 + l15;
    const _Float16* pA = sc + h * (NPAD * NPAD);
    v8f acc = {};
#pragma unroll
    for (int kk = 0; kk < 2; ++kk) {
      v16h a  = load_a_f16(pA, m, NPAD, kk * 32, lane);
      v16h bf = load_b_rowmajor16(vst, col, VSTRIDE, kk * 32, lane);
      acc = wmma_f16(a, bf, acc);
    }
#pragma unroll
    for (int r = 0; r < 8; ++r) {
      const int mr = tm * 16 + r + rbase;
      xs[mr * DIMC + col] = (_Float16)acc[r];
    }
  }
  __syncthreads();

  // ---- Phase 6: projection (64x128 @ 128x128) + bias -> global -------------
  for (int i = 0; i < 4; ++i) {
    const int t  = wv * 4 + i;
    const int tm = t & 3;
    const int tn = t >> 2;               // 0..7
    const int m  = tm * 16 + l15;
    const int n  = tn * 16 + l15;
    v8f acc = {};
#pragma unroll
    for (int kk = 0; kk < 4; ++kk) {
      v16h a  = load_a_f16(xs, m, DIMC, kk * 32, lane);
      v16h bf = load_b_rowmajor(projw, n, DIMC, kk * 32, lane);
      acc = wmma_f16(a, bf, acc);
    }
    const float pb = proj_b[n];
    const size_t obase = (size_t)b * (NTOK * DIMC);
#pragma unroll
    for (int r = 0; r < 8; ++r) {
      const int mr = tm * 16 + r + rbase;
      if (mr < NTOK) out[obase + mr * DIMC + n] = acc[r] + pb;
    }
  }
}

// ---------------------------------------------------------------------------

extern "C" void kernel_launch(void* const* d_in, const int* in_sizes, int n_in,
                              void* d_out, int out_size, void* d_ws, size_t ws_size,
                              hipStream_t stream) {
  const float* x      = (const float*)d_in[0];
  const float* mask   = (const float*)d_in[1];
  const float* qkv_w  = (const float*)d_in[2];
  const float* qkv_b  = (const float*)d_in[3];
  const float* proj_w = (const float*)d_in[4];
  const float* proj_b = (const float*)d_in[5];
  const float* rpb    = (const float*)d_in[6];
  const int*   rpi    = (const int*)d_in[7];

  char* ws = (char*)d_ws;
  _Float16* qkvw_h  = (_Float16*)(ws);                // 384*128*2 = 98304 B
  _Float16* projw_h = (_Float16*)(ws + 98304);        // 128*128*2 = 32768 B
  float*    bias    = (float*)(ws + 98304 + 32768);   // 4*49*49*4 = 38416 B

  // Prep: f16 weights + gathered relative-position bias.
  swin_prep_kernel<<<(THREE_C * DIMC + 255) / 256, 256, 0, stream>>>(
      qkv_w, proj_w, rpb, rpi, qkvw_h, projw_h, bias);

  // Main: one workgroup per window, ~98 KB dynamic LDS.
  const size_t lds_bytes = LDS_TOTAL * sizeof(_Float16);
  swin_attn_kernel<<<4096, 256, lds_bytes, stream>>>(
      x, mask, qkv_b, proj_b, qkvw_h, projw_h, bias, (float*)d_out);
}
